// ResiQ1FF_41042707481000
// MI455X (gfx1250) — compile-verified, matrix operand order checked by reading, not run on hardware
//
#include <hip/hip_runtime.h>

// ---------------------------------------------------------------------------
// Types
// ---------------------------------------------------------------------------
typedef __bf16 bf16;
typedef __bf16 v16bf __attribute__((ext_vector_type(16)));
typedef float  v8f   __attribute__((ext_vector_type(8)));
typedef int    v4i   __attribute__((ext_vector_type(4)));

typedef __attribute__((address_space(1))) v4i gas_v4i;  // global
typedef __attribute__((address_space(3))) v4i las_v4i;  // LDS

struct bf16x4 { bf16 a, b, c, d; };
struct FragBits { uint4 lo, hi; };   // 32 bytes == 16 bf16 == one WMMA A/B fragment per lane

__device__ __forceinline__ bf16 f2bf(float f) {
  unsigned u = __builtin_bit_cast(unsigned, f);
  u += 0x7FFFu + ((u >> 16) & 1u);                 // round-to-nearest-even
  unsigned short h = (unsigned short)(u >> 16);
  return __builtin_bit_cast(bf16, h);
}
__device__ __forceinline__ float bf2f(bf16 v) {
  unsigned u = ((unsigned)__builtin_bit_cast(unsigned short, v)) << 16;
  return __builtin_bit_cast(float, u);
}

// HARDNESS = 0.95 -> 100/(1.025-0.95) = 1333.333
__device__ __forceinline__ float soft_sign_f(float w) {
  float s = 1.0f / (1.0f + __expf(-w * 1333.3333f));
  return 0.05f * w + 0.95f * (2.0f * s - 1.0f);
}
__device__ __forceinline__ float soft_bin_f(float w) {
  float s = 1.0f / (1.0f + __expf(-w * 1333.3333f));
  return 0.05f * w + 0.95f * s;
}

// ---------------------------------------------------------------------------
// gfx1250 async global->LDS staging (ASYNCcnt path), with sync fallback
// ---------------------------------------------------------------------------
#if defined(__AMDGCN__) && __has_builtin(__builtin_amdgcn_global_load_async_to_lds_b128)
#define USE_ASYNC_LDS 1
#else
#define USE_ASYNC_LDS 0
#endif

__device__ __forceinline__ void stage_b128(const bf16* g, bf16* l) {
#if USE_ASYNC_LDS
  __builtin_amdgcn_global_load_async_to_lds_b128(
      (gas_v4i*)(uintptr_t)g,
      (las_v4i*)(unsigned)(uintptr_t)l,
      0, 0);
#else
  *(uint4*)l = *(const uint4*)g;
#endif
}

__device__ __forceinline__ void stage_wait() {
#if USE_ASYNC_LDS
#if __has_builtin(__builtin_amdgcn_s_wait_asynccnt)
  __builtin_amdgcn_s_wait_asynccnt(0);
#else
  asm volatile("s_wait_asynccnt 0x0" ::: "memory");
#endif
#endif
}

// ---------------------------------------------------------------------------
// Elementwise prep kernels
// ---------------------------------------------------------------------------
__global__ __launch_bounds__(256) void k_cvt_bf16(const float* __restrict__ x,
                                                  bf16* __restrict__ o,
                                                  long long n4) {
  long long i = (long long)blockIdx.x * 256 + threadIdx.x;
  if (i < n4) {
    float4 v = ((const float4*)x)[i];
    bf16x4 r{f2bf(v.x), f2bf(v.y), f2bf(v.z), f2bf(v.w)};
    ((bf16x4*)o)[i] = r;
  }
}

// W (K x N, fp32, row-major)  ->  out (N x K, bf16, row-major) with soft op.
// mode 0 = soft_sign, 1 = soft_bin.  LDS-tiled transpose, coalesced both sides.
__global__ __launch_bounds__(256) void k_prep_wt(const float* __restrict__ W,
                                                 bf16* __restrict__ out,
                                                 int K, int N, int mode) {
  __shared__ float t[32][33];
  const int k0 = blockIdx.y * 32, n0 = blockIdx.x * 32;
  const int tx = threadIdx.x, ty = threadIdx.y;   // block (32, 8)
#pragma unroll
  for (int i = 0; i < 4; ++i) {
    int k = ty + i * 8;
    t[k][tx] = W[(size_t)(k0 + k) * N + (n0 + tx)];
  }
  __syncthreads();
#pragma unroll
  for (int i = 0; i < 4; ++i) {
    int n = ty + i * 8;
    float w = t[tx][n];
    float v = (mode == 0) ? soft_sign_f(w) : soft_bin_f(w);
    out[(size_t)(n0 + n) * K + (k0 + tx)] = f2bf(v);
  }
}

__global__ __launch_bounds__(256) void k_prep_w3(const float* __restrict__ W3,
                                                 float* __restrict__ o, int n) {
  int i = blockIdx.x * 256 + threadIdx.x;
  if (i < n) o[i] = soft_bin_f(W3[i]);
}

// ---------------------------------------------------------------------------
// bf16 WMMA GEMM:  out[M,N] = relu(A[M,K] * Bt[N,K]^T * s[n] + bia[n]) (+resid)
// Block tile 128x128, 8 waves (4x2), wave tile 32x64 (2x4 WMMA frags).
// Async double-buffered LDS staging, one barrier per K-step.
// ---------------------------------------------------------------------------
#define BM 128
#define BN 128
#define BK 32
#define LP 40   // LDS row pitch in bf16 elements (80B) -> conflict-free b128 frag reads

__global__ __launch_bounds__(256) void k_gemm_bf16(
    const bf16* __restrict__ A,    // [M][K]
    const bf16* __restrict__ Bt,   // [N][K]
    const float* __restrict__ s,   // [N]
    const float* __restrict__ bia, // [N]
    const bf16* __restrict__ resid,// [M][N] or nullptr
    float* __restrict__ out,       // [M][N]
    int M, int N, int K) {
  __shared__ __align__(16) bf16 lA[2][BM * LP];
  __shared__ __align__(16) bf16 lB[2][BN * LP];

  const int tid   = threadIdx.x;
  const int lane  = tid & 31;
  const int wave  = tid >> 5;
  const int waveM = wave & 3;    // 0..3  (rows of 32)
  const int waveN = wave >> 2;   // 0..1  (cols of 64)
  const int m0 = blockIdx.y * BM;
  const int n0 = blockIdx.x * BN;

  const int fr = lane & 15;      // fragment row within 16
  const int k4 = lane >> 4;      // which 16B K-chunk pair (ISA 16-bit A/B layout)

  // Staging: 256 threads move 2x(128x32) bf16 per tile = 4 x b128 per thread.
  const int rowS = tid >> 2;           // 0..63
  const int colS = (tid & 3) * 8;      // 0,8,16,24

  v8f acc[2][4];
#pragma unroll
  for (int mi = 0; mi < 2; ++mi)
#pragma unroll
    for (int ni = 0; ni < 4; ++ni)
#pragma unroll
      for (int e = 0; e < 8; ++e) acc[mi][ni][e] = 0.0f;

  auto stage_tile = [&](int buf, int kk) {
    stage_b128(&A[(size_t)(m0 + rowS) * K + kk + colS],       &lA[buf][rowS * LP + colS]);
    stage_b128(&A[(size_t)(m0 + rowS + 64) * K + kk + colS],  &lA[buf][(rowS + 64) * LP + colS]);
    stage_b128(&Bt[(size_t)(n0 + rowS) * K + kk + colS],      &lB[buf][rowS * LP + colS]);
    stage_b128(&Bt[(size_t)(n0 + rowS + 64) * K + kk + colS], &lB[buf][(rowS + 64) * LP + colS]);
  };

  stage_tile(0, 0);

  int buf = 0;
  for (int kk = 0; kk < K; kk += BK) {
    stage_wait();        // our async tile writes are done (ASYNCcnt==0)
    __syncthreads();     // everyone's are done; prev readers of buf^1 finished

    if (kk + BK < K) stage_tile(buf ^ 1, kk + BK);   // prefetch next tile async

    // Load fragments (two ds_load_b128 each; banks spaced 4 apart mod 64)
    v16bf aF[2], bF[4];
#pragma unroll
    for (int mi = 0; mi < 2; ++mi) {
      const uint4* p = (const uint4*)&lA[buf][(waveM * 32 + mi * 16 + fr) * LP];
      FragBits f{p[k4], p[k4 + 2]};
      aF[mi] = __builtin_bit_cast(v16bf, f);
    }
#pragma unroll
    for (int ni = 0; ni < 4; ++ni) {
      const uint4* p = (const uint4*)&lB[buf][(waveN * 64 + ni * 16 + fr) * LP];
      FragBits f{p[k4], p[k4 + 2]};
      bF[ni] = __builtin_bit_cast(v16bf, f);
    }

#pragma unroll
    for (int mi = 0; mi < 2; ++mi)
#pragma unroll
      for (int ni = 0; ni < 4; ++ni)
        acc[mi][ni] = __builtin_amdgcn_wmma_f32_16x16x32_bf16(
            false, aF[mi], false, bF[ni], (short)0, acc[mi][ni], false, false);

    buf ^= 1;
  }

  // Epilogue: *s + b, relu, optional residual add, store fp32.
#pragma unroll
  for (int mi = 0; mi < 2; ++mi) {
    const int mBase = m0 + waveM * 32 + mi * 16 + k4 * 8;
#pragma unroll
    for (int ni = 0; ni < 4; ++ni) {
      const int n = n0 + waveN * 64 + ni * 16 + fr;
      const float sv = s[n];
      const float bv = bia[n];
#pragma unroll
      for (int e = 0; e < 8; ++e) {
        const int m = mBase + e;
        float v = fmaf(acc[mi][ni][e], sv, bv);
        v = fmaxf(v, 0.0f);
        if (resid) v += bf2f(resid[(size_t)m * N + n]);
        out[(size_t)m * N + n] = v;
      }
    }
  }
}

// ---------------------------------------------------------------------------
// Row LayerNorm over N=8192 fp32 -> bf16 (g, be applied)
// ---------------------------------------------------------------------------
__global__ __launch_bounds__(256) void k_layernorm(const float* __restrict__ y,
                                                   const float* __restrict__ g,
                                                   const float* __restrict__ be,
                                                   bf16* __restrict__ o, int N) {
  const int row = blockIdx.x, tid = threadIdx.x;
  const float* yr = y + (size_t)row * N;
  float s = 0.0f, ss = 0.0f;
  for (int i = tid * 4; i < N; i += 1024) {
    float4 v = *(const float4*)(yr + i);
    s  += v.x + v.y + v.z + v.w;
    ss += v.x * v.x + v.y * v.y + v.z * v.z + v.w * v.w;
  }
  __shared__ float rs[256], rq[256];
  rs[tid] = s; rq[tid] = ss;
  __syncthreads();
  for (int k = 128; k > 0; k >>= 1) {
    if (tid < k) { rs[tid] += rs[tid + k]; rq[tid] += rq[tid + k]; }
    __syncthreads();
  }
  const float mean = rs[0] / (float)N;
  const float var  = rq[0] / (float)N - mean * mean;
  const float rstd = rsqrtf(var + 1e-12f);
  bf16* orow = o + (size_t)row * N;
  for (int i = tid * 4; i < N; i += 1024) {
    float4 v  = *(const float4*)(yr + i);
    float4 gv = *(const float4*)(g + i);
    float4 bv = *(const float4*)(be + i);
    bf16x4 r{f2bf((v.x - mean) * rstd * gv.x + bv.x),
             f2bf((v.y - mean) * rstd * gv.y + bv.y),
             f2bf((v.z - mean) * rstd * gv.z + bv.z),
             f2bf((v.w - mean) * rstd * gv.w + bv.w)};
    *(bf16x4*)(orow + i) = r;
  }
}

// ---------------------------------------------------------------------------
// Head: out[m][j] = sum_k x2[m][k] * w3s[k][j] * s3[j] + b3[j]   (j < 10)
// ---------------------------------------------------------------------------
__global__ __launch_bounds__(256) void k_head(const bf16* __restrict__ x2,
                                              const float* __restrict__ w3s,
                                              const float* __restrict__ s3,
                                              const float* __restrict__ b3,
                                              float* __restrict__ out,
                                              int K, int DOUT) {
  const int m = blockIdx.x, tid = threadIdx.x;
  float acc[10];
#pragma unroll
  for (int j = 0; j < 10; ++j) acc[j] = 0.0f;
  const bf16* xr = x2 + (size_t)m * K;
  for (int k = tid; k < K; k += 256) {
    float xv = bf2f(xr[k]);
    const float* wr = w3s + (size_t)k * 10;
#pragma unroll
    for (int j = 0; j < 10; ++j) acc[j] = fmaf(xv, wr[j], acc[j]);
  }
  __shared__ float red[10][256];
#pragma unroll
  for (int j = 0; j < 10; ++j) red[j][tid] = acc[j];
  __syncthreads();
  for (int k = 128; k > 0; k >>= 1) {
    if (tid < k)
#pragma unroll
      for (int j = 0; j < 10; ++j) red[j][tid] += red[j][tid + k];
    __syncthreads();
  }
  if (tid < DOUT) out[(size_t)m * DOUT + tid] = red[tid][0] * s3[tid] + b3[tid];
}

// ---------------------------------------------------------------------------
// Launch
// ---------------------------------------------------------------------------
extern "C" void kernel_launch(void* const* d_in, const int* in_sizes, int n_in,
                              void* d_out, int out_size, void* d_ws, size_t ws_size,
                              hipStream_t stream) {
  const float* x   = (const float*)d_in[0];
  const float* W1  = (const float*)d_in[1];
  const float* b1  = (const float*)d_in[2];
  const float* s1  = (const float*)d_in[3];
  const float* g1  = (const float*)d_in[4];
  const float* be1 = (const float*)d_in[5];
  const float* W2  = (const float*)d_in[6];
  const float* b2  = (const float*)d_in[7];
  const float* s2  = (const float*)d_in[8];
  const float* g2  = (const float*)d_in[9];
  const float* be2 = (const float*)d_in[10];
  const float* W3  = (const float*)d_in[11];
  const float* b3  = (const float*)d_in[12];
  const float* s3  = (const float*)d_in[13];
  float* out = (float*)d_out;

  constexpr int Bb = 8192, DIN = 3072, H = 8192, DOUT = 10;

  char* ws = (char*)d_ws;
  bf16*  xbf  = (bf16*)(ws);                                   //  50331648 B
  bf16*  w1t  = (bf16*)(ws + (size_t)50331648);                //  50331648 B  [H][DIN]
  bf16*  w2t  = (bf16*)(ws + (size_t)100663296);               // 134217728 B  [H][H]
  float* ybuf = (float*)(ws + (size_t)234881024);              // 268435456 B  (reused)
  bf16*  x1bf = (bf16*)(ws + (size_t)503316480);               // 134217728 B
  bf16*  x2bf = (bf16*)(ws + (size_t)637534208);               // 134217728 B
  float* w3s  = (float*)(ws + (size_t)771751936);              //    327680 B

  // prep
  k_cvt_bf16<<<(Bb * (long long)DIN / 4 + 255) / 256, 256, 0, stream>>>(
      x, xbf, (long long)Bb * DIN / 4);
  k_prep_wt<<<dim3(H / 32, DIN / 32), dim3(32, 8), 0, stream>>>(W1, w1t, DIN, H, 0);
  k_prep_wt<<<dim3(H / 32, H / 32), dim3(32, 8), 0, stream>>>(W2, w2t, H, H, 1);
  k_prep_w3<<<(H * DOUT + 255) / 256, 256, 0, stream>>>(W3, w3s, H * DOUT);

  // layer 1: GEMM + scale/bias/relu fused, then LN -> bf16
  k_gemm_bf16<<<dim3(H / BN, Bb / BM), 256, 0, stream>>>(
      xbf, w1t, s1, b1, nullptr, ybuf, Bb, H, DIN);
  k_layernorm<<<Bb, 256, 0, stream>>>(ybuf, g1, be1, x1bf, H);

  // layer 2: GEMM + scale/bias/relu + residual fused, then LN -> bf16
  k_gemm_bf16<<<dim3(H / BN, Bb / BM), 256, 0, stream>>>(
      x1bf, w2t, s2, b2, x1bf, ybuf, Bb, H, H);
  k_layernorm<<<Bb, 256, 0, stream>>>(ybuf, g2, be2, x2bf, H);

  // head
  k_head<<<Bb, 256, 0, stream>>>(x2bf, w3s, s3, b3, out, H, DOUT);
}
